// Generator_429496729630
// MI455X (gfx1250) — matrix operand order repaired and checked by reading, not compiled, not run on hardware
//
#include <hip/hip_runtime.h>
#include <hip/hip_bf16.h>

typedef _Float16 f16;
typedef __attribute__((ext_vector_type(16))) _Float16 v16h;
typedef __attribute__((ext_vector_type(8)))  float    v8f;
typedef __attribute__((ext_vector_type(4)))  int      v4i;

#define B_      256
#define T_      64
#define EMB_    128
#define HID_    512
#define COMBIN_ 704
#define GATES_  2048
#define VTOT_   3072

#if defined(__has_builtin)
#if __has_builtin(__builtin_amdgcn_global_load_async_to_lds_b128) && \
    __has_builtin(__builtin_amdgcn_s_wait_asynccnt)
#define USE_ASYNC_LDS 1
#endif
#endif

#ifdef USE_ASYNC_LDS
// b128 async copy: (global int4* src, lds int4* dst, imm offset, imm cpol)
#define ASYNC_COPY16(dst_lds, src_glb)                                           \
  __builtin_amdgcn_global_load_async_to_lds_b128(                                \
      (__attribute__((address_space(1))) v4i*)(src_glb),                         \
      (__attribute__((address_space(3))) v4i*)(dst_lds), 0, 0)
#define ASYNC_WAIT() __builtin_amdgcn_s_wait_asynccnt(0)
#else
#define ASYNC_COPY16(dst_lds, src_glb) \
  (*(uint4*)(dst_lds) = *(const uint4*)(src_glb))
#define ASYNC_WAIT() ((void)0)
#endif

// ---------------- WMMA fragment helpers (CDNA5 16x16x32 f16 layouts) ----------------

// A-matrix 16x32 (MxK), row-major source with row stride `lda` (elements).
__device__ __forceinline__ v16h load_a_frag(const f16* row0, int lda, int lane) {
  int m = lane & 15, hi = lane >> 4;
  const f16* r = row0 + (size_t)m * lda;
  v16h a;
#pragma unroll
  for (int v = 0; v < 8; ++v) {
    int k = ((v >> 2) << 4) + (hi << 3) + ((v & 3) << 1);
    a[2 * v]     = r[k];
    a[2 * v + 1] = r[k + 1];
  }
  return a;
}

// B-matrix 32x16 (KxN) where B[k][n] = W[n][k], W row-major with row stride ldw.
__device__ __forceinline__ v16h load_b_fragW(const f16* w0, int ldw, int lane) {
  int n = lane & 15, hi = lane >> 4;
  const f16* r = w0 + (size_t)n * ldw;
  v16h b;
#pragma unroll
  for (int v = 0; v < 8; ++v) {
    int k = ((v >> 2) << 4) + (hi << 3) + ((v & 3) << 1);
    b[2 * v]     = r[k];
    b[2 * v + 1] = r[k + 1];
  }
  return b;
}

__device__ __forceinline__ v8f wmma16(v16h a, v16h b, v8f c) {
  return __builtin_amdgcn_wmma_f32_16x16x32_f16(false, a, false, b, (short)0, c,
                                                false, false);
}

// ---------------- prep kernels ----------------

__global__ void k_cvt_f16(const float* __restrict__ s, f16* __restrict__ d, int n) {
  for (int i = blockIdx.x * blockDim.x + threadIdx.x; i < n; i += gridDim.x * blockDim.x)
    d[i] = (f16)s[i];
}

// stacked LSTM weight: dst[l][n][k], k<512 from Wih, k>=512 from Whh
__global__ void k_build_lstmW(const float* __restrict__ Wih, const float* __restrict__ Whh,
                              f16* __restrict__ dst) {
  const int total = 4 * GATES_ * 1024;
  for (int i = blockIdx.x * blockDim.x + threadIdx.x; i < total; i += gridDim.x * blockDim.x) {
    int l = i >> 21;
    int rem = i & ((1 << 21) - 1);
    int n = rem >> 10;
    int k = rem & 1023;
    float v = (k < HID_) ? Wih[((size_t)l * GATES_ + n) * HID_ + k]
                         : Whh[((size_t)l * GATES_ + n) * HID_ + (k - HID_)];
    dst[i] = (f16)v;
  }
}

__global__ void k_build_lstmBias(const float* __restrict__ bih, const float* __restrict__ bhh,
                                 float* __restrict__ dst) {
  int i = blockIdx.x * blockDim.x + threadIdx.x;
  if (i < 4 * GATES_) dst[i] = bih[i] + bhh[i];
}

__global__ void k_build_outW(const float* __restrict__ w0, const float* __restrict__ w1,
                             const float* __restrict__ w2, const float* __restrict__ w3,
                             const float* __restrict__ w4, f16* __restrict__ dst) {
  const int total = VTOT_ * 576;
  for (int i = blockIdx.x * blockDim.x + threadIdx.x; i < total; i += gridDim.x * blockDim.x) {
    int r = i / 576, k = i - r * 576;
    const float* src; int lr;
    if (r < 256)       { src = w0; lr = r; }
    else if (r < 512)  { src = w1; lr = r - 256; }
    else if (r < 1536) { src = w2; lr = r - 512; }
    else if (r < 2560) { src = w3; lr = r - 1536; }
    else               { src = w4; lr = r - 2560; }
    dst[i] = (f16)src[(size_t)lr * 576 + k];
  }
}

__global__ void k_build_outB(const float* __restrict__ b0, const float* __restrict__ b1,
                             const float* __restrict__ b2, const float* __restrict__ b3,
                             const float* __restrict__ b4, float* __restrict__ dst) {
  int r = blockIdx.x * blockDim.x + threadIdx.x;
  if (r >= VTOT_) return;
  if (r < 256)       dst[r] = b0[r];
  else if (r < 512)  dst[r] = b1[r - 256];
  else if (r < 1536) dst[r] = b2[r - 512];
  else if (r < 2560) dst[r] = b3[r - 1536];
  else               dst[r] = b4[r - 2560];
}

__global__ void k_label_argmax(const float* __restrict__ label, int* __restrict__ li) {
  int b = blockIdx.x * blockDim.x + threadIdx.x;
  if (b >= B_) return;
  const float* p = label + (size_t)b * 8;
  int best = 0; float bv = p[0];
#pragma unroll
  for (int j = 1; j < 8; ++j) if (p[j] > bv) { bv = p[j]; best = j; }
  li[b] = best;
}

__global__ void k_length(const int* __restrict__ length, const float* __restrict__ lenW,
                         const float* __restrict__ lenB, float* __restrict__ out) {
  int i = blockIdx.x * blockDim.x + threadIdx.x;
  if (i >= B_ * 64) return;
  int b = i >> 6, j = i & 63;
  int li = length[b] - 1;
  li = li < 0 ? 0 : (li > 63 ? 63 : li);
  float v = lenW[j * 64 + li] + lenB[j];
  out[i] = v > 0.f ? v : 0.f;
}

__global__ void k_combined(const int* __restrict__ x,
                           const float* __restrict__ e0, const float* __restrict__ e1,
                           const float* __restrict__ e2, const float* __restrict__ e3,
                           const float* __restrict__ e4,
                           const float* __restrict__ lenout, f16* __restrict__ dst) {
  const int total = B_ * T_ * COMBIN_;
  for (int i = blockIdx.x * blockDim.x + threadIdx.x; i < total; i += gridDim.x * blockDim.x) {
    int row = i / COMBIN_;
    int col = i - row * COMBIN_;
    float v;
    if (col < 640) {
      int a = col >> 7, kk = col & 127;
      const float* e = (a == 0) ? e0 : (a == 1) ? e1 : (a == 2) ? e2 : (a == 3) ? e3 : e4;
      v = e[(size_t)x[(size_t)row * 5 + a] * EMB_ + kk];
    } else {
      v = lenout[(row >> 6) * 64 + (col - 640)];
    }
    dst[i] = (f16)v;
  }
}

__global__ void k_attr0(const float* __restrict__ pad, f16* __restrict__ attr) {
  int i = blockIdx.x * blockDim.x + threadIdx.x;
  if (i >= B_ * T_ * 64) return;
  int row = i >> 6, k = i & 63;
  attr[((size_t)row * 5) * 64 + k] = (f16)pad[k];
}

__global__ void k_cond(const int* __restrict__ x_now,
                       const float* __restrict__ e0, const float* __restrict__ e1,
                       const float* __restrict__ e2, const float* __restrict__ e3,
                       const float* __restrict__ w0, const float* __restrict__ w1,
                       const float* __restrict__ w2, const float* __restrict__ w3,
                       const float* __restrict__ b0, const float* __restrict__ b1,
                       const float* __restrict__ b2, const float* __restrict__ b3,
                       f16* __restrict__ attr) {
  const int total = B_ * T_ * 4 * 64;
  for (int i = blockIdx.x * blockDim.x + threadIdx.x; i < total; i += gridDim.x * blockDim.x) {
    int j = i & 63;
    int h = (i >> 6) & 3;
    int row = i >> 8;
    int K = (h + 1) * EMB_;
    const float* W  = (h == 0) ? w0 : (h == 1) ? w1 : (h == 2) ? w2 : w3;
    const float* bb = (h == 0) ? b0 : (h == 1) ? b1 : (h == 2) ? b2 : b3;
    const float* Wr = W + (size_t)j * K;
    float acc = bb[j];
    for (int c = 0; c <= h; ++c) {
      const float* e  = (c == 0) ? e0 : (c == 1) ? e1 : (c == 2) ? e2 : e3;
      const float* er = e + (size_t)x_now[(size_t)row * 5 + c] * EMB_;
      const float* wr = Wr + c * EMB_;
#pragma unroll 4
      for (int kk = 0; kk < EMB_; ++kk) acc += er[kk] * wr[kk];
    }
    acc = acc > 0.f ? acc : 0.f;
    attr[((size_t)row * 5 + (h + 1)) * 64 + j] = (f16)acc;
  }
}

// ---------------- combine_fc: [64x704]x[704x512] per batch row, label-routed --------------
// block = 256 thr (8 waves) computes a 16Mx512N slab; wave -> 16x64 (4 accumulators).
// grid = B * 4 (M tiles)
__global__ void k_combine_fc(const f16* __restrict__ comb, const f16* __restrict__ cw,
                             const float* __restrict__ cb, const int* __restrict__ li,
                             f16* __restrict__ xout) {
  __shared__ f16 As[16 * COMBIN_];  // 22 KB
  int tid = threadIdx.x;
  int b  = blockIdx.x >> 2;
  int mt = blockIdx.x & 3;
  int L = li[b];
  // stage A tile: 16 rows x 704 halves = 1408 16B chunks (88 per row)
  for (int c = tid; c < 1408; c += 256) {
    int row = c / 88;
    int k0 = (c - row * 88) << 3;
    ASYNC_COPY16(&As[row * COMBIN_ + k0],
                 comb + (size_t)(b * T_ + mt * 16 + row) * COMBIN_ + k0);
  }
  ASYNC_WAIT();
  __syncthreads();
  int wv = tid >> 5, lane = tid & 31;
  int n0 = wv * 64;  // 8 waves cover N=512
  const f16* W0 = cw + ((size_t)L * HID_ + n0) * COMBIN_;
  v8f a0 = {}, a1 = {}, a2 = {}, a3 = {};
  for (int k0 = 0; k0 < COMBIN_; k0 += 32) {
    v16h a = load_a_frag(&As[k0], COMBIN_, lane);
    a0 = wmma16(a, load_b_fragW(W0 + k0, COMBIN_, lane), a0);
    a1 = wmma16(a, load_b_fragW(W0 + (size_t)16 * COMBIN_ + k0, COMBIN_, lane), a1);
    a2 = wmma16(a, load_b_fragW(W0 + (size_t)32 * COMBIN_ + k0, COMBIN_, lane), a2);
    a3 = wmma16(a, load_b_fragW(W0 + (size_t)48 * COMBIN_ + k0, COMBIN_, lane), a3);
  }
  int n = lane & 15, hi = lane >> 4;
  const v8f* accs[4] = {&a0, &a1, &a2, &a3};
#pragma unroll
  for (int j = 0; j < 4; ++j) {
    float bias = cb[(size_t)L * HID_ + n0 + 16 * j + n];
#pragma unroll
    for (int v = 0; v < 8; ++v) {
      int m = hi * 8 + v;
      float val = (*accs[j])[v] + bias;
      val = val > 0.f ? val : 0.f;
      xout[(size_t)(b * T_ + mt * 16 + m) * HID_ + n0 + 16 * j + n] = (f16)val;
    }
  }
}

// ---------------- LSTM recurrent step: g = [x_t | h_{t-1}] * [Wih|Whh]^T + bias ----------
// block = 256 thr (8 waves): M tile (16 batch rows) x 512 gate cols; wave -> 16x64.
// grid = 16 (M tiles) * 4 (N groups) = 64 blocks
__global__ void k_lstm_gemm(const f16* __restrict__ xin, const f16* __restrict__ h16,
                            const f16* __restrict__ W, const float* __restrict__ bias,
                            float* __restrict__ g, int t) {
  __shared__ f16 As[16 * 1024];  // 32 KB
  int tid = threadIdx.x;
  int mt  = blockIdx.x >> 2;
  int grp = blockIdx.x & 3;
  // stage A tile: k<512 from x_t, k>=512 from h_{t-1}; 2048 16B chunks (128/row)
  for (int c = tid; c < 2048; c += 256) {
    int row = c >> 7;
    int k0 = (c & 127) << 3;
    const f16* src = (k0 < 512)
        ? xin + ((size_t)(mt * 16 + row) * T_ + t) * HID_ + k0
        : h16 + (size_t)(mt * 16 + row) * HID_ + (k0 - 512);
    ASYNC_COPY16(&As[row * 1024 + k0], src);
  }
  ASYNC_WAIT();
  __syncthreads();
  int wv = tid >> 5, lane = tid & 31;
  int n0 = (grp * 8 + wv) * 64;  // 0..1984
  const f16* W0 = W + (size_t)n0 * 1024;
  v8f a0 = {}, a1 = {}, a2 = {}, a3 = {};
  for (int k0 = 0; k0 < 1024; k0 += 32) {
    __builtin_prefetch(W0 + k0 + 1024, 0, 1);  // global_prefetch_b8 on weight stream
    v16h a = load_a_frag(&As[k0], 1024, lane);
    a0 = wmma16(a, load_b_fragW(W0 + k0, 1024, lane), a0);
    a1 = wmma16(a, load_b_fragW(W0 + (size_t)16 * 1024 + k0, 1024, lane), a1);
    a2 = wmma16(a, load_b_fragW(W0 + (size_t)32 * 1024 + k0, 1024, lane), a2);
    a3 = wmma16(a, load_b_fragW(W0 + (size_t)48 * 1024 + k0, 1024, lane), a3);
  }
  int n = lane & 15, hi = lane >> 4;
  const v8f* accs[4] = {&a0, &a1, &a2, &a3};
#pragma unroll
  for (int j = 0; j < 4; ++j) {
    float bb = bias[n0 + 16 * j + n];
#pragma unroll
    for (int v = 0; v < 8; ++v) {
      int m = hi * 8 + v;
      g[(size_t)(mt * 16 + m) * GATES_ + n0 + 16 * j + n] = (*accs[j])[v] + bb;
    }
  }
}

__global__ void k_zero_state(float* __restrict__ c, f16* __restrict__ h) {
  int i = blockIdx.x * blockDim.x + threadIdx.x;
  if (i < B_ * HID_) { c[i] = 0.f; h[i] = (f16)0.f; }
}

// gates (torch order i,f,g,o): c' = sig(f)*c + sig(i)*tanh(g); h = sig(o)*tanh(c')
__global__ void k_lstm_gate(const float* __restrict__ g, float* __restrict__ c,
                            f16* __restrict__ h16, f16* __restrict__ xout, int t) {
  int i = blockIdx.x * blockDim.x + threadIdx.x;
  if (i >= B_ * HID_) return;
  int b = i >> 9, j = i & 511;
  const float* gr = g + (size_t)b * GATES_;
  float gi = gr[j], gf = gr[HID_ + j], gg = gr[2 * HID_ + j], go = gr[3 * HID_ + j];
  float si = 1.f / (1.f + __expf(-gi));
  float sf = 1.f / (1.f + __expf(-gf));
  float so = 1.f / (1.f + __expf(-go));
  float cn = sf * c[i] + si * tanhf(gg);
  float hh = so * tanhf(cn);
  c[i] = cn;
  h16[i] = (f16)hh;
  xout[((size_t)(b * T_) + t) * HID_ + j] = (f16)hh;
}

// ---------------- output heads: [16384x576]x[576x3072] -> d_out (preactivations) ----------
// block = 256 thr (8 waves): 16M x 512N slab; wave -> 16x64 (single head per wave).
// grid = 1024 (M tiles) * 6 (N groups) = 6144 blocks
__global__ void k_head_gemm(const f16* __restrict__ hidden, const f16* __restrict__ attr,
                            const f16* __restrict__ W, const float* __restrict__ bias,
                            float* __restrict__ out) {
  __shared__ f16 Ah[16 * 512];  // 16 KB hidden
  __shared__ f16 Aa[16 * 320];  // 10 KB all 5 attr heads
  int tid = threadIdx.x;
  int mt  = blockIdx.x / 6;
  int grp = blockIdx.x - mt * 6;
  for (int c = tid; c < 1024; c += 256) {  // 16*512/8 chunks (64/row)
    int row = c >> 6;
    int k0 = (c & 63) << 3;
    ASYNC_COPY16(&Ah[row * 512 + k0], hidden + (size_t)(mt * 16 + row) * HID_ + k0);
  }
  for (int c = tid; c < 640; c += 256) {   // 16*320/8 chunks (40/row)
    int row = c / 40;
    int k0 = (c - row * 40) << 3;
    ASYNC_COPY16(&Aa[row * 320 + k0], attr + (size_t)(mt * 16 + row) * 320 + k0);
  }
  ASYNC_WAIT();
  __syncthreads();
  int wv = tid >> 5, lane = tid & 31;
  int n0 = (grp * 8 + wv) * 64;  // head boundaries (256/512/1536/2560) are 64-aligned
  int head = (n0 < 256) ? 0 : (n0 < 512) ? 1 : (n0 < 1536) ? 2 : (n0 < 2560) ? 3 : 4;
  const f16* W0 = W + (size_t)n0 * 576;
  v8f a0 = {}, a1 = {}, a2 = {}, a3 = {};
  for (int k0 = 0; k0 < HID_; k0 += 32) {
    v16h a = load_a_frag(&Ah[k0], 512, lane);
    a0 = wmma16(a, load_b_fragW(W0 + k0, 576, lane), a0);
    a1 = wmma16(a, load_b_fragW(W0 + (size_t)16 * 576 + k0, 576, lane), a1);
    a2 = wmma16(a, load_b_fragW(W0 + (size_t)32 * 576 + k0, 576, lane), a2);
    a3 = wmma16(a, load_b_fragW(W0 + (size_t)48 * 576 + k0, 576, lane), a3);
  }
  for (int k0 = 0; k0 < 64; k0 += 32) {
    v16h a = load_a_frag(&Aa[head * 64 + k0], 320, lane);
    a0 = wmma16(a, load_b_fragW(W0 + 512 + k0, 576, lane), a0);
    a1 = wmma16(a, load_b_fragW(W0 + (size_t)16 * 576 + 512 + k0, 576, lane), a1);
    a2 = wmma16(a, load_b_fragW(W0 + (size_t)32 * 576 + 512 + k0, 576, lane), a2);
    a3 = wmma16(a, load_b_fragW(W0 + (size_t)48 * 576 + 512 + k0, 576, lane), a3);
  }
  int n = lane & 15, hi = lane >> 4;
  const v8f* accs[4] = {&a0, &a1, &a2, &a3};
#pragma unroll
  for (int j = 0; j < 4; ++j) {
    float bb = bias[n0 + 16 * j + n];
#pragma unroll
    for (int v = 0; v < 8; ++v) {
      int m = hi * 8 + v;
      out[(size_t)(mt * 16 + m) * VTOT_ + n0 + 16 * j + n] = (*accs[j])[v] + bb;
    }
  }
}

// in-place log-softmax over each head segment of each row; one block per (row, head)
__global__ void k_logsoftmax(float* __restrict__ out) {
  __shared__ float sm[256];
  const int off[5] = {0, 256, 512, 1536, 2560};
  const int sz[5]  = {256, 256, 1024, 1024, 512};
  int row = blockIdx.x / 5, h = blockIdx.x % 5;
  float* p = out + (size_t)row * VTOT_ + off[h];
  int V = sz[h];
  int tid = threadIdx.x;
  float m = -1e30f;
  for (int i = tid; i < V; i += 256) m = fmaxf(m, p[i]);
  sm[tid] = m; __syncthreads();
  for (int s = 128; s > 0; s >>= 1) { if (tid < s) sm[tid] = fmaxf(sm[tid], sm[tid + s]); __syncthreads(); }
  m = sm[0]; __syncthreads();
  float acc = 0.f;
  for (int i = tid; i < V; i += 256) acc += __expf(p[i] - m);
  sm[tid] = acc; __syncthreads();
  for (int s = 128; s > 0; s >>= 1) { if (tid < s) sm[tid] += sm[tid + s]; __syncthreads(); }
  float lse = m + __logf(sm[0]); __syncthreads();
  for (int i = tid; i < V; i += 256) p[i] = p[i] - lse;
}

// ---------------- host orchestration ----------------

extern "C" void kernel_launch(void* const* d_in, const int* in_sizes, int n_in,
                              void* d_out, int out_size, void* d_ws, size_t ws_size,
                              hipStream_t stream) {
  (void)in_sizes; (void)n_in; (void)out_size; (void)ws_size;
  const float* label   = (const float*)d_in[0];
  const int*   length  = (const int*)  d_in[1];
  const int*   x       = (const int*)  d_in[2];
  const int*   x_now   = (const int*)  d_in[3];
  const float* emb0    = (const float*)d_in[4];
  const float* emb1    = (const float*)d_in[5];
  const float* emb2    = (const float*)d_in[6];
  const float* emb3    = (const float*)d_in[7];
  const float* emb4    = (const float*)d_in[8];
  const float* pad_emb = (const float*)d_in[9];
  const float* cond_W[4] = {(const float*)d_in[10], (const float*)d_in[12],
                            (const float*)d_in[14], (const float*)d_in[16]};
  const float* cond_b[4] = {(const float*)d_in[11], (const float*)d_in[13],
                            (const float*)d_in[15], (const float*)d_in[17]};
  const float* len_W   = (const float*)d_in[18];
  const float* len_b   = (const float*)d_in[19];
  const float* comb_W  = (const float*)d_in[20];
  const float* comb_b  = (const float*)d_in[21];
  const float* lstm_Wih = (const float*)d_in[22];
  const float* lstm_Whh = (const float*)d_in[23];
  const float* lstm_bih = (const float*)d_in[24];
  const float* lstm_bhh = (const float*)d_in[25];
  const float* out_W[5] = {(const float*)d_in[26], (const float*)d_in[28],
                           (const float*)d_in[30], (const float*)d_in[32],
                           (const float*)d_in[34]};
  const float* out_b[5] = {(const float*)d_in[27], (const float*)d_in[29],
                           (const float*)d_in[31], (const float*)d_in[33],
                           (const float*)d_in[35]};
  float* out = (float*)d_out;

  // workspace carve (~97 MB total)
  char* p = (char*)d_ws;
  auto alloc = [&](size_t bytes) -> void* {
    void* r = (void*)p;
    p += (bytes + 255) & ~((size_t)255);
    return r;
  };
  f16*   combW16  = (f16*)  alloc((size_t)8 * HID_ * COMBIN_ * sizeof(f16));
  f16*   lstmW16  = (f16*)  alloc((size_t)4 * GATES_ * 1024 * sizeof(f16));
  float* lstmBias = (float*)alloc((size_t)4 * GATES_ * sizeof(float));
  f16*   outW16   = (f16*)  alloc((size_t)VTOT_ * 576 * sizeof(f16));
  float* outB     = (float*)alloc((size_t)VTOT_ * sizeof(float));
  int*   labelInt = (int*)  alloc((size_t)B_ * sizeof(int));
  float* lenOut   = (float*)alloc((size_t)B_ * 64 * sizeof(float));
  f16*   comb16   = (f16*)  alloc((size_t)B_ * T_ * COMBIN_ * sizeof(f16));
  f16*   attr16   = (f16*)  alloc((size_t)B_ * T_ * 5 * 64 * sizeof(f16));
  f16*   xA       = (f16*)  alloc((size_t)B_ * T_ * HID_ * sizeof(f16));
  f16*   xB       = (f16*)  alloc((size_t)B_ * T_ * HID_ * sizeof(f16));
  float* gbuf     = (float*)alloc((size_t)B_ * GATES_ * sizeof(float));
  float* cbuf     = (float*)alloc((size_t)B_ * HID_ * sizeof(float));
  f16*   h16      = (f16*)  alloc((size_t)B_ * HID_ * sizeof(f16));

  const int TB = 256;
  {
    int n = 8 * HID_ * COMBIN_;
    k_cvt_f16<<<(n + TB - 1) / TB, TB, 0, stream>>>(comb_W, combW16, n);
  }
  k_build_lstmW<<<8192, TB, 0, stream>>>(lstm_Wih, lstm_Whh, lstmW16);
  k_build_lstmBias<<<(4 * GATES_ + TB - 1) / TB, TB, 0, stream>>>(lstm_bih, lstm_bhh, lstmBias);
  k_build_outW<<<4096, TB, 0, stream>>>(out_W[0], out_W[1], out_W[2], out_W[3], out_W[4], outW16);
  k_build_outB<<<(VTOT_ + TB - 1) / TB, TB, 0, stream>>>(out_b[0], out_b[1], out_b[2], out_b[3],
                                                         out_b[4], outB);
  k_label_argmax<<<1, TB, 0, stream>>>(label, labelInt);
  k_length<<<(B_ * 64 + TB - 1) / TB, TB, 0, stream>>>(length, len_W, len_b, lenOut);
  k_combined<<<8192, TB, 0, stream>>>(x, emb0, emb1, emb2, emb3, emb4, lenOut, comb16);
  k_attr0<<<(B_ * T_ * 64 + TB - 1) / TB, TB, 0, stream>>>(pad_emb, attr16);
  k_cond<<<8192, TB, 0, stream>>>(x_now, emb0, emb1, emb2, emb3,
                                  cond_W[0], cond_W[1], cond_W[2], cond_W[3],
                                  cond_b[0], cond_b[1], cond_b[2], cond_b[3], attr16);

  // combine_fc (WMMA, LDS-staged A): grid = B * 4 M-tiles
  k_combine_fc<<<B_ * 4, TB, 0, stream>>>(comb16, combW16, comb_b, labelInt, xA);

  // 4-layer LSTM: per (layer, t): fused GEMM (64 blocks) + gates
  for (int l = 0; l < 4; ++l) {
    const f16* xin  = (l & 1) ? xB : xA;
    f16*       xout = (l & 1) ? xA : xB;
    const f16*   Wl = lstmW16 + (size_t)l * GATES_ * 1024;
    const float* bl = lstmBias + (size_t)l * GATES_;
    k_zero_state<<<(B_ * HID_ + TB - 1) / TB, TB, 0, stream>>>(cbuf, h16);
    for (int t = 0; t < T_; ++t) {
      k_lstm_gemm<<<64, TB, 0, stream>>>(xin, h16, Wl, bl, gbuf, t);
      k_lstm_gate<<<(B_ * HID_ + TB - 1) / TB, TB, 0, stream>>>(gbuf, cbuf, h16, xout, t);
    }
  }
  const f16* hidden = xA;  // layer 3 writes xA

  // heads (WMMA, LDS-staged A): grid = 1024 M-tiles * 6 N-groups
  k_head_gemm<<<1024 * 6, TB, 0, stream>>>(hidden, attr16, outW16, outB, out);

  // log-softmax per (row, head)
  k_logsoftmax<<<B_ * T_ * 5, TB, 0, stream>>>(out);
}
